// DecoderHead_6408091206311
// MI455X (gfx1250) — matrix operand order repaired and checked by reading, not compiled, not run on hardware
//
#include <hip/hip_runtime.h>
#include <cstdint>

// Problem constants (from reference)
constexpr int B_ = 8;
constexpr int T_ = 2048;
constexpr int C_ = 512;
constexpr int H_ = 512;

typedef __attribute__((ext_vector_type(16))) __bf16    v16bf;
typedef __attribute__((ext_vector_type(8)))  float     v8f;
typedef __attribute__((ext_vector_type(4)))  unsigned  uint4v;

union Frag {
    v16bf    v;
    unsigned u[8];
    uint4v   q[2];
};

union BfPair {
    __bf16   h[2];
    unsigned u;
};

__device__ __forceinline__ v8f wmma_bf16(const v16bf& a, const v16bf& b, const v8f& c) {
    return __builtin_amdgcn_wmma_f32_16x16x32_bf16(false, a, false, b, (short)0, c,
                                                   false, false);
}

// ---------------------------------------------------------------------------
// Kernel 0: one-time f32 -> bf16 conversion (used for the three weights).
// ---------------------------------------------------------------------------
__global__ __launch_bounds__(256) void cvt_bf16_kernel(const float* __restrict__ src,
                                                       __bf16* __restrict__ dst, int n) {
    int i = blockIdx.x * blockDim.x + threadIdx.x;
    const int stride = gridDim.x * blockDim.x;
    for (; i < n; i += stride) dst[i] = (__bf16)src[i];
}

// ---------------------------------------------------------------------------
// Kernel 1: QKV projection.  Each wave computes a 16(M=t) x 64(N=h) group for
// Q, K and V at once: the x A-fragment is loaded once per K-step and reused by
// 12 WMMAs (4 N-tiles x 3 matrices).  Weights are pre-converted bf16, loaded
// as contiguous 32B B-fragments.  Q is pre-scaled by 1/sqrt(H).  K stored
// [B*T][H] bf16, V stored transposed [B][H][T] bf16 so attention B-fragments
// are contiguous.
// ---------------------------------------------------------------------------
__global__ __launch_bounds__(256) void proj_qkv_kernel(
    const float*  __restrict__ x,    // [B*T][C] f32
    const __bf16* __restrict__ Wqb,  // [H][C] bf16
    const __bf16* __restrict__ Wkb,  // [H][C] bf16
    const __bf16* __restrict__ Wvb,  // [H][C] bf16
    __bf16* __restrict__ Qb,         // [B*T][H]
    __bf16* __restrict__ Kb,         // [B*T][H]
    __bf16* __restrict__ Vt)         // [B][H][T]
{
    const int lane = threadIdx.x & 31;
    const int wave = threadIdx.x >> 5;
    const int hh   = lane >> 4;    // lane half
    const int ln   = lane & 15;

    const int NG   = H_ / 64;                       // 8 N-groups
    const int tile = blockIdx.x * 8 + wave;
    const int mt   = tile / NG;                     // token tile (0..1023) — shared by block
    const int ng   = tile % NG;                     // 64-wide head-dim group (= wave id)

    const int m = mt * 16 + ln;                     // flattened b*T + t row

    v8f acc[12];                                    // [mat(q,k,v)][nt2]
#pragma unroll
    for (int i = 0; i < 12; ++i) acc[i] = v8f{};

    for (int c0 = 0; c0 < C_; c0 += 32) {
        // A fragment from f32 x: two contiguous 8-float runs per lane.
        const float* xr = x + (size_t)m * C_ + c0;
        const float* x0 = xr + 16 * hh;             // K = {0..7}  + 16*hh
        const float* x1 = xr + 16 * hh + 8;         // K = {8..15} + 16*hh
        Frag a;
#pragma unroll
        for (int e = 0; e < 8; ++e) {
            a.v[e]     = (__bf16)x0[e];
            a.v[8 + e] = (__bf16)x1[e];
        }
#pragma unroll
        for (int nt2 = 0; nt2 < 4; ++nt2) {
            const int n = ng * 64 + nt2 * 16 + ln;  // head-dim column for this lane
            const size_t woff = (size_t)n * C_ + c0 + 16 * hh;   // 32B contiguous
            Frag bq, bk, bv;
            const uint4v* q0 = (const uint4v*)(Wqb + woff);
            const uint4v* k0 = (const uint4v*)(Wkb + woff);
            const uint4v* v0 = (const uint4v*)(Wvb + woff);
            bq.q[0] = q0[0]; bq.q[1] = q0[1];
            bk.q[0] = k0[0]; bk.q[1] = k0[1];
            bv.q[0] = v0[0]; bv.q[1] = v0[1];
            acc[0 + nt2] = wmma_bf16(a.v, bq.v, acc[0 + nt2]);
            acc[4 + nt2] = wmma_bf16(a.v, bk.v, acc[4 + nt2]);
            acc[8 + nt2] = wmma_bf16(a.v, bv.v, acc[8 + nt2]);
        }
    }

    const float qscale = 0.044194173824159216f;     // 1/sqrt(512)
#pragma unroll
    for (int nt2 = 0; nt2 < 4; ++nt2) {
        const int col = ng * 64 + nt2 * 16 + ln;
#pragma unroll
        for (int r = 0; r < 8; ++r) {
            const int row = mt * 16 + r + 8 * hh;   // token index (flattened)
            Qb[(size_t)row * H_ + col] = (__bf16)(acc[0 + nt2][r] * qscale);
            Kb[(size_t)row * H_ + col] = (__bf16)(acc[4 + nt2][r]);
            const int b = row / T_;
            const int t = row % T_;
            Vt[((size_t)b * H_ + col) * T_ + t] = (__bf16)(acc[8 + nt2][r]);
        }
    }
}

// ---------------------------------------------------------------------------
// Kernel 2: flash attention.  One 256-thread block (8 waves) per
// (batch, 16-query tile).  Waves split H: wave w owns head dims [64w,64w+64).
// Per 32-key step: each wave computes its partial S = Q_chunk*K_chunk^T
// (4 WMMAs), partials reduced into LDS via ds_add_f32, wave0 runs the online
// softmax with all 32 lanes (halves combined by shfl_xor), then all waves
// rescale O and do P*V (4 WMMAs) for their own 64 output dims.
// ---------------------------------------------------------------------------
__global__ __launch_bounds__(256) void flash_attn_kernel(
    const __bf16* __restrict__ Qb,  // [B*T][H]
    const __bf16* __restrict__ Kb,  // [B*T][H]
    const __bf16* __restrict__ Vt,  // [B][H][T]
    float* __restrict__ out)        // [B][T][H]
{
    __shared__ __align__(16) float  S[16][32];       // reduced scores
    __shared__ __align__(16) __bf16 P[16][32];       // probabilities (bf16)
    __shared__ float m_s[16], l_s[16], alpha_s[16];

    const int lane = threadIdx.x & 31;
    const int wave = threadIdx.x >> 5;
    const int hh   = lane >> 4;
    const int ln   = lane & 15;

    const int qt = blockIdx.x;          // query tile
    const int b  = blockIdx.y;          // batch
    const int t0 = qt * 16;
    const int hd0 = wave * 64;          // this wave's head-dim chunk
    const size_t baseTH = (size_t)b * T_ * H_;

    if (threadIdx.x < 16) { m_s[threadIdx.x] = -__builtin_inff(); l_s[threadIdx.x] = 0.f; }

    // Cache Q A-fragments for this wave's 64-dim chunk: 2 x (two b128 loads).
    Frag qf[2];
#pragma unroll
    for (int f = 0; f < 2; ++f) {
        const __bf16* qrow = Qb + baseTH + (size_t)(t0 + ln) * H_ + hd0 + f * 32;
        qf[f].q[0] = *(const uint4v*)(qrow + 16 * hh);       // K {0..7}+16hh
        qf[f].q[1] = *(const uint4v*)(qrow + 16 * hh + 8);   // K {8..15}+16hh
    }

    v8f O[4] = {v8f{}, v8f{}, v8f{}, v8f{}};

    for (int s0 = 0; s0 < t0 + 16; s0 += 32) {
        const bool haveNext = (s0 + 32) < (t0 + 16);

        // ---- phase a: zero the reduced-score tile --------------------------
        float* Sf = &S[0][0];
        Sf[threadIdx.x]       = 0.f;
        Sf[threadIdx.x + 256] = 0.f;
        __syncthreads();

        // ---- phase b: partial S = Q_chunk * K_chunk^T, reduce into LDS -----
#pragma unroll
        for (int ntile = 0; ntile < 2; ++ntile) {
            if (haveNext)   // prefetch next step's K rows (L2 latency hiding)
                __builtin_prefetch(Kb + baseTH +
                    (size_t)(s0 + 32 + ntile * 16 + ln) * H_ + hd0, 0, 3);
            v8f acc = {};
#pragma unroll
            for (int f = 0; f < 2; ++f) {
                Frag kf;   // B-fragment: lane = key column, 16 contiguous hd K
                const uint4v* src = (const uint4v*)(Kb + baseTH +
                    (size_t)(s0 + ntile * 16 + ln) * H_ + hd0 + f * 32 + 16 * hh);
                kf.q[0] = src[0];
                kf.q[1] = src[1];
                acc = wmma_bf16(qf[f].v, kf.v, acc);
            }
#pragma unroll
            for (int r = 0; r < 8; ++r)
                atomicAdd(&S[r + 8 * hh][ntile * 16 + ln], acc[r]);
        }
        __syncthreads();

        // ---- phase c: wave0 online softmax; lane = (row=ln, col-half=hh) ---
        if (wave == 0) {
            const int row = ln;
            const int t   = t0 + row;
            const float mo = m_s[row];

            float val[16];
            float mxl = -__builtin_inff();
#pragma unroll
            for (int j = 0; j < 16; ++j) {
                const int s = s0 + 16 * hh + j;
                const bool valid = (s <= t);
                val[j] = valid ? S[row][16 * hh + j] : -__builtin_inff();
                if (valid) mxl = fmaxf(mxl, val[j]);
            }
            const float mx   = fmaxf(mxl, __shfl_xor(mxl, 16, 32));
            const float mnew = fmaxf(mo, mx);
            const float a    = __expf(mo - mnew);    // 0 on first step

            float suml = 0.f;
            unsigned* Prow = (unsigned*)&P[row][16 * hh];
#pragma unroll
            for (int j2 = 0; j2 < 8; ++j2) {
                const int s = s0 + 16 * hh + 2 * j2;
                const float p0 = (s     <= t) ? __expf(val[2 * j2]     - mnew) : 0.f;
                const float p1 = (s + 1 <= t) ? __expf(val[2 * j2 + 1] - mnew) : 0.f;
                BfPair pk;
                pk.h[0] = (__bf16)p0;
                pk.h[1] = (__bf16)p1;
                Prow[j2] = pk.u;
                suml += p0 + p1;
            }
            const float sum = suml + __shfl_xor(suml, 16, 32);
            if (hh == 0) {
                m_s[row]     = mnew;
                alpha_s[row] = a;
                l_s[row]     = l_s[row] * a + sum;
            }
        }
        __syncthreads();

        // ---- phase d: rescale O, accumulate O += P * V_chunk ---------------
        float ar[8];
#pragma unroll
        for (int r = 0; r < 8; ++r) ar[r] = alpha_s[r + 8 * hh];

        Frag pf;   // A-fragment of P from LDS: two b128 reads per lane
        pf.q[0] = *(const uint4v*)&P[ln][16 * hh];
        pf.q[1] = *(const uint4v*)&P[ln][16 * hh + 8];

#pragma unroll
        for (int nt2 = 0; nt2 < 4; ++nt2) {
            const size_t vrow = ((size_t)b * H_ + hd0 + nt2 * 16 + ln) * T_;
            if (haveNext)   // prefetch next step's V keys
                __builtin_prefetch(Vt + vrow + s0 + 32, 0, 3);

            Frag vf;   // B-fragment: lane = output h, 16 contiguous keys (Vt)
            const uint4v* vsrc = (const uint4v*)(Vt + vrow + s0 + 16 * hh);
            vf.q[0] = vsrc[0];
            vf.q[1] = vsrc[1];

            v8f o = O[nt2];
#pragma unroll
            for (int r = 0; r < 8; ++r) o[r] *= ar[r];
            O[nt2] = wmma_bf16(pf.v, vf.v, o);
        }
    }
    __syncthreads();

    // ---- epilogue: normalize by l and store f32 ----------------------------
#pragma unroll
    for (int nt2 = 0; nt2 < 4; ++nt2) {
        const int col = hd0 + nt2 * 16 + ln;
#pragma unroll
        for (int r = 0; r < 8; ++r) {
            const int row  = t0 + r + 8 * hh;
            const float li = 1.f / l_s[r + 8 * hh];
            out[baseTH + (size_t)row * H_ + col] = O[nt2][r] * li;
        }
    }
}

// ---------------------------------------------------------------------------
extern "C" void kernel_launch(void* const* d_in, const int* in_sizes, int n_in,
                              void* d_out, int out_size, void* d_ws, size_t ws_size,
                              hipStream_t stream) {
    // setup_inputs() order: x, y, z, mask, Wk, Wq, Wv   (y, z, mask unused;
    // mask is causal tril and is applied analytically).
    const float* x  = (const float*)d_in[0];
    const float* Wk = (const float*)d_in[4];
    const float* Wq = (const float*)d_in[5];
    const float* Wv = (const float*)d_in[6];
    float* out = (float*)d_out;

    const size_t nTH = (size_t)B_ * T_ * H_;   // 8.4M elements
    const size_t nW  = (size_t)H_ * C_;        // 256K elements
    __bf16* Qb  = (__bf16*)d_ws;               // 16 MB
    __bf16* Kb  = Qb + nTH;                    // 16 MB
    __bf16* Vt  = Kb + nTH;                    // 16 MB
    __bf16* Wqb = Vt + nTH;                    // 0.5 MB
    __bf16* Wkb = Wqb + nW;                    // 0.5 MB
    __bf16* Wvb = Wkb + nW;                    // 0.5 MB   (total ~49.5 MB)

    // One-time weight conversion to bf16.
    cvt_bf16_kernel<<<256, 256, 0, stream>>>(Wq, Wqb, (int)nW);
    cvt_bf16_kernel<<<256, 256, 0, stream>>>(Wk, Wkb, (int)nW);
    cvt_bf16_kernel<<<256, 256, 0, stream>>>(Wv, Wvb, (int)nW);

    // Projection: (B*T/16) M-tiles x (H/64) N-groups, 8 waves per block
    // (all waves of a block share the same x rows).
    const int projBlocks = (B_ * T_ / 16) * (H_ / 64) / 8;   // 1024
    proj_qkv_kernel<<<projBlocks, 256, 0, stream>>>(x, Wqb, Wkb, Wvb, Qb, Kb, Vt);

    // Flash attention: one block per (query tile, batch).
    flash_attn_kernel<<<dim3(T_ / 16, B_), 256, 0, stream>>>(Qb, Kb, Vt, out);
}
